// LaneLoss_26259430048055
// MI455X (gfx1250) — compile-verified, matrix operand order checked by reading, not run on hardware
//
#include <hip/hip_runtime.h>
#include <hip/hip_bf16.h>
#include <math.h>

// Problem constants (from reference)
#define BB 256
#define QQ 50
#define LL 16
#define NN 72
#define EL 144   // N*2 flattened per (q,l) pair
#define TT 7
#define EXIST_W 2.0f
#define PTS_W   5.0f
#define TYPE_W  1.0f
#define POS_W   3.0f

typedef __attribute__((ext_vector_type(16))) _Float16     v16h;
typedef __attribute__((ext_vector_type(8)))  float        v8f;
typedef __attribute__((ext_vector_type(8)))  unsigned int v8u;

// Scalar |x| via integer bit ops (avoids the buggy packed-f32 fabs combine).
__device__ __forceinline__ float absf(float x) {
    return __uint_as_float(__float_as_uint(x) & 0x7fffffffu);
}

// ---------------------------------------------------------------------------
// Kernel 1: cost[b][l][q] = mean|pred_pts[b,q]-gt_points[b,l]| - 0.5*sigmoid(pe[b,q])
// 256 blocks x 128 threads (4 waves). Wave w handles q-rows [16w,16w+16).
// The 144-element L1 reduction is done with v_wmma_f32_16x16x32_f16 against an
// all-ones B matrix: D = A x 1  => every D column holds the row-sum, so one
// WMMA reduces 16 rows x 32 elements. Signed diffs are converted to f16 and
// the sign bits are cleared with an integer mask (|(f16)d| == (f16)|d|).
// ---------------------------------------------------------------------------
__global__ void __launch_bounds__(128)
lane_cost_kernel(const float* __restrict__ pe,   // (B,Q,1)
                 const float* __restrict__ pp,   // (B,Q,N,2)
                 const float* __restrict__ gp,   // (B,L,N,2)
                 float* __restrict__ costout)    // (B,L,Q)
{
    __shared__ float spp[QQ * EL];   // 28800 B
    __shared__ float sgp[LL * EL];   //  9216 B
    __shared__ float ssig[QQ];

    const int b = blockIdx.x;
    const int t = threadIdx.x;

    const float* ppb = pp + (size_t)b * QQ * EL;
    const float* gpb = gp + (size_t)b * LL * EL;
    for (int i = t; i < QQ * EL; i += 128) spp[i] = ppb[i];
    for (int i = t; i < LL * EL; i += 128) sgp[i] = gpb[i];
    if (t < QQ) ssig[t] = 1.0f / (1.0f + expf(-pe[b * QQ + t]));
    __syncthreads();

    const int wave  = t >> 5;
    const int lane  = t & 31;
    const int qbase = wave * 16;                 // waves 0..3 cover q 0..63
    const int row   = qbase + (lane & 15);
    const int rq    = row < QQ ? row : (QQ - 1); // clamp: keep EXEC uniform
    const int kbase = (lane >> 4) * 8;           // f16 A-matrix lane layout

    v16h bones;
    #pragma unroll
    for (int e = 0; e < 16; ++e) bones[e] = (_Float16)1.0f;

    for (int l = 0; l < LL; ++l) {
        v8f acc = {0.f, 0.f, 0.f, 0.f, 0.f, 0.f, 0.f, 0.f};
        #pragma unroll
        for (int c = 0; c < 5; ++c) {            // 5 chunks of K=32 cover 144
            v16h a;
            #pragma unroll
            for (int e = 0; e < 16; ++e) {
                // ISA 16-bit A layout: VGPR0..3 -> K=e, VGPR4..7 -> K=e+8 (+8 for hi lanes)
                int kk  = 32 * c + kbase + e + ((e >= 8) ? 8 : 0);
                int kks = kk < EL ? kk : 0;
                float d = spp[rq * EL + kks] - sgp[l * EL + kks];
                d = (kk < EL) ? d : 0.0f;
                a[e] = (_Float16)d;              // signed; sign cleared below
            }
            // |a| elementwise: clear f16 sign bits with an integer mask.
            v8u au = __builtin_bit_cast(v8u, a);
            au = au & 0x7fff7fffu;
            a = __builtin_bit_cast(v16h, au);

            acc = __builtin_amdgcn_wmma_f32_16x16x32_f16(
                false, a, false, bones, (short)0, acc, false, false);
        }
        // D columns are identical (ones-B): lanes 0..15 hold rows 0..7 in acc[0..7],
        // lanes 16..31 hold rows 8..15. Let lane 0 / 16 write 8 rows each.
        if (lane == 0 || lane == 16) {
            int rbase = qbase + ((lane >> 4) ? 8 : 0);
            #pragma unroll
            for (int r = 0; r < 8; ++r) {
                int q = rbase + r;
                if (q < QQ)
                    costout[((size_t)b * LL + l) * QQ + q] =
                        acc[r] * (1.0f / (float)EL) - 0.5f * ssig[q];
            }
        }
    }
}

// ---------------------------------------------------------------------------
// Kernel 2: per-sample Hungarian (e-maxx potentials), one wave32 per batch.
// Rows = existing GT lanes (n<=16), cols = 50 queries. Column work (minv
// update, argmin) is wave-parallel; the scalar j0-chain lives in LDS.
// ---------------------------------------------------------------------------
__global__ void __launch_bounds__(32)
hungarian_kernel(const float* __restrict__ cost,  // (B,L,Q)
                 const float* __restrict__ ge,    // (B,L)
                 const float* __restrict__ hl,    // (B,)
                 int* __restrict__ nmatch,        // (B,)
                 int* __restrict__ mq,            // (B,16)
                 int* __restrict__ mg)            // (B,16)
{
    const int b = blockIdx.x;
    const int lane = threadIdx.x;
    const int M = QQ;   // 50 columns

    __shared__ float cst[LL][QQ];
    __shared__ float u[LL + 1];
    __shared__ float v[QQ + 1];
    __shared__ float minv[QQ + 1];
    __shared__ int   p[QQ + 1];
    __shared__ int   way[QQ + 1];
    __shared__ int   used[QQ + 1];
    __shared__ int   gidx[LL];
    __shared__ int   sh_n, sh_j0;

    if (lane == 0) {
        int n = 0;
        for (int l = 0; l < LL; ++l)
            if (ge[b * LL + l] > 0.5f) gidx[n++] = l;
        sh_n = (hl[b] > 0.5f) ? n : 0;
    }
    __syncthreads();
    const int n = sh_n;
    if (lane == 0) nmatch[b] = n;
    if (n == 0) return;

    for (int k = lane; k < n * QQ; k += 32) {
        int i = k / QQ, j = k % QQ;
        cst[i][j] = cost[((size_t)b * LL + gidx[i]) * QQ + j];
    }
    for (int j = lane; j <= M; j += 32) { v[j] = 0.f; p[j] = 0; way[j] = 0; }
    for (int i = lane; i <= LL; i += 32) u[i] = 0.f;
    __syncthreads();

    for (int i = 1; i <= n; ++i) {
        if (lane == 0) { p[0] = i; sh_j0 = 0; }
        for (int j = lane; j <= M; j += 32) { minv[j] = INFINITY; used[j] = 0; }
        __syncthreads();

        while (true) {
            int j0 = sh_j0;
            if (lane == 0) used[j0] = 1;
            __syncthreads();
            int   i0  = p[j0];
            float ui0 = u[i0];
            // update minv over free columns
            for (int j = 1 + lane; j <= M; j += 32) {
                if (!used[j]) {
                    float cur = cst[i0 - 1][j - 1] - ui0 - v[j];
                    if (cur < minv[j]) { minv[j] = cur; way[j] = j0; }
                }
            }
            __syncthreads();
            // argmin over free columns (first-index tie-break, like np.argmin)
            float bv = INFINITY; int bj = M + 1;
            for (int j = 1 + lane; j <= M; j += 32)
                if (!used[j] && minv[j] < bv) { bv = minv[j]; bj = j; }
            #pragma unroll
            for (int off = 16; off > 0; off >>= 1) {
                float ov = __shfl_xor(bv, off, 32);
                int   oj = __shfl_xor(bj, off, 32);
                if (ov < bv || (ov == bv && oj < bj)) { bv = ov; bj = oj; }
            }
            const float delta = bv;
            const int   j1    = bj;
            // potential / minv updates (used columns map to distinct rows -> no race)
            for (int j = lane; j <= M; j += 32) {
                if (used[j]) { u[p[j]] += delta; v[j] -= delta; }
                else if (j >= 1) minv[j] -= delta;
            }
            if (lane == 0) sh_j0 = j1;
            __syncthreads();
            if (p[sh_j0] == 0) break;
        }
        __syncthreads();
        if (lane == 0) {                 // augment along 'way' chain
            int j0 = sh_j0;
            while (j0 != 0) { int j1 = way[j0]; p[j0] = p[j1]; j0 = j1; }
        }
        __syncthreads();
    }
    // emit: row (p[j]-1) matched to column (j-1)
    for (int j = 1 + lane; j <= M; j += 32) {
        int r = p[j];
        if (r != 0) {
            mq[b * LL + (r - 1)] = j - 1;
            mg[b * LL + (r - 1)] = gidx[r - 1];
        }
    }
}

// ---------------------------------------------------------------------------
// Kernel 3: per-batch loss partials (deterministic per-batch ordering).
// 256 blocks x 128 threads (4 waves; each wave handles whole matches for pts).
// ---------------------------------------------------------------------------
__device__ __forceinline__ float softplus_f(float x) {
    return fmaxf(x, 0.0f) + log1pf(expf(-absf(x)));
}

__global__ void __launch_bounds__(128)
loss_partial_kernel(const float* __restrict__ pe,    // (B,Q,1)
                    const float* __restrict__ pp,    // (B,Q,N,2)
                    const float* __restrict__ ptype, // (B,Q,T)
                    const float* __restrict__ gp,    // (B,L,N,2)
                    const float* __restrict__ gv,    // (B,L,N)
                    const int*   __restrict__ glt,   // (B,L)
                    const float* __restrict__ hl,    // (B,)
                    const int*   __restrict__ nmatch,
                    const int*   __restrict__ mq,
                    const int*   __restrict__ mg,
                    float* __restrict__ bexist,
                    float* __restrict__ bpts,
                    float* __restrict__ btype,
                    int*   __restrict__ bn,
                    int*   __restrict__ bflag)
{
    const int b = blockIdx.x;
    const int t = threadIdx.x;
    const int lane = t & 31, wave = t >> 5;

    __shared__ int   s_n, s_mq[LL], s_mg[LL];
    __shared__ unsigned char s_t[QQ + 2];
    __shared__ float e_terms[QQ];
    __shared__ float pts_val[LL];
    __shared__ float type_val[LL];

    if (t == 0) s_n = nmatch[b];
    if (t < LL) { s_mq[t] = mq[b * LL + t]; s_mg[t] = mg[b * LL + t];
                  pts_val[t] = 0.f; type_val[t] = 0.f; }
    if (t < QQ + 2) s_t[t] = 0;
    __syncthreads();
    const int n = s_n;
    const int flag = (hl[b] > 0.5f) ? 1 : 0;
    if (t < n) s_t[s_mq[t]] = 1;
    __syncthreads();

    // existence BCE terms (one per query)
    if (t < QQ) {
        float x  = pe[b * QQ + t];
        float tt = s_t[t] ? 1.0f : 0.0f;
        e_terms[t] = POS_W * tt * softplus_f(-x) + (1.0f - tt) * softplus_f(x);
    }
    // type cross-entropy (one thread per match, T=7 serial)
    if (t < n) {
        const float* lg = ptype + ((size_t)b * QQ + s_mq[t]) * TT;
        int tg = glt[b * LL + s_mg[t]];
        float m = lg[0];
        #pragma unroll
        for (int k = 1; k < TT; ++k) m = fmaxf(m, lg[k]);
        float s = 0.f;
        #pragma unroll
        for (int k = 0; k < TT; ++k) s += expf(lg[k] - m);
        type_val[t] = -(lg[tg] - m - logf(s));
    }
    // smooth-L1 points loss: wave `wave` handles matches wave, wave+4, ...
    for (int i = wave; i < n; i += 4) {
        int q = s_mq[i], g = s_mg[i];
        const float* P = pp + ((size_t)b * QQ + q) * EL;
        const float* G = gp + ((size_t)b * LL + g) * EL;
        const float* V = gv + ((size_t)b * LL + g) * NN;
        float num = 0.f;
        for (int e = lane; e < EL; e += 32) {
            float d = P[e] - G[e];
            float ad = absf(d);
            float sl1 = (ad < 1.0f) ? 0.5f * d * d : ad - 0.5f;
            num += sl1 * V[e >> 1];
        }
        float den = 0.f;
        for (int nn = lane; nn < NN; nn += 32) den += V[nn];
        #pragma unroll
        for (int off = 16; off > 0; off >>= 1) {
            num += __shfl_xor(num, off, 32);
            den += __shfl_xor(den, off, 32);
        }
        if (lane == 0) pts_val[i] = num / fmaxf(den, 1.0f);
    }
    __syncthreads();

    if (t == 0) {   // fixed-order per-batch sums -> deterministic
        float es = 0.f;
        if (flag) for (int q = 0; q < QQ; ++q) es += e_terms[q];
        float ps = 0.f, ts = 0.f;
        for (int i = 0; i < n; ++i) { ps += pts_val[i]; ts += type_val[i]; }
        bexist[b] = es; bpts[b] = ps; btype[b] = ts; bn[b] = n; bflag[b] = flag;
    }
}

// ---------------------------------------------------------------------------
// Kernel 4: single-block fixed-order reduction -> final scalar.
// ---------------------------------------------------------------------------
__global__ void __launch_bounds__(256)
loss_final_kernel(const float* __restrict__ bexist,
                  const float* __restrict__ bpts,
                  const float* __restrict__ btype,
                  const int*   __restrict__ bn,
                  const int*   __restrict__ bflag,
                  float* __restrict__ out)
{
    __shared__ float se[BB], sp_[BB], st[BB];
    __shared__ int   sn[BB], sf[BB];
    const int t = threadIdx.x;
    se[t] = bexist[t]; sp_[t] = bpts[t]; st[t] = btype[t];
    sn[t] = bn[t];     sf[t]  = bflag[t];
    __syncthreads();
    for (int s = 128; s > 0; s >>= 1) {   // fixed tree -> deterministic
        if (t < s) {
            se[t]  += se[t + s];  sp_[t] += sp_[t + s];
            st[t]  += st[t + s];  sn[t]  += sn[t + s];  sf[t] += sf[t + s];
        }
        __syncthreads();
    }
    if (t == 0) {
        int nb = sf[0];
        if (nb == 0) { out[0] = 0.0f; return; }
        float exist_loss = se[0] / ((float)nb * (float)QQ);
        int nm = sn[0] > 0 ? sn[0] : 1;
        out[0] = EXIST_W * exist_loss
               + PTS_W  * sp_[0] / (float)nm
               + TYPE_W * st[0]  / (float)nm;
    }
}

// ---------------------------------------------------------------------------
extern "C" void kernel_launch(void* const* d_in, const int* in_sizes, int n_in,
                              void* d_out, int out_size, void* d_ws, size_t ws_size,
                              hipStream_t stream) {
    const float* pe    = (const float*)d_in[0]; // pred_exist   (B,Q,1)
    const float* pp    = (const float*)d_in[1]; // pred_pts     (B,Q,N,2)
    const float* ptype = (const float*)d_in[2]; // pred_type    (B,Q,T)
    const float* ge    = (const float*)d_in[3]; // gt_existence (B,L)
    const float* gp    = (const float*)d_in[4]; // gt_points    (B,L,N,2)
    const float* gv    = (const float*)d_in[5]; // gt_visibility(B,L,N)
    const int*   glt   = (const int*)  d_in[6]; // gt_lane_type (B,L)
    const float* hl    = (const float*)d_in[7]; // has_lanes    (B,)

    // Workspace carve-up (~858 KB total)
    float* cost   = (float*)d_ws;                        // B*L*Q
    int*   nmatch = (int*)(cost + (size_t)BB * LL * QQ); // B
    int*   mq     = nmatch + BB;                         // B*16
    int*   mg     = mq + BB * LL;                        // B*16
    float* bexist = (float*)(mg + BB * LL);              // B
    float* bpts   = bexist + BB;                         // B
    float* btype  = bpts + BB;                           // B
    int*   bn     = (int*)(btype + BB);                  // B
    int*   bflag  = bn + BB;                             // B

    lane_cost_kernel<<<BB, 128, 0, stream>>>(pe, pp, gp, cost);
    hungarian_kernel<<<BB, 32, 0, stream>>>(cost, ge, hl, nmatch, mq, mg);
    loss_partial_kernel<<<BB, 128, 0, stream>>>(pe, pp, ptype, gp, gv, glt, hl,
                                                nmatch, mq, mg,
                                                bexist, bpts, btype, bn, bflag);
    loss_final_kernel<<<1, BB, 0, stream>>>(bexist, bpts, btype, bn, bflag,
                                            (float*)d_out);
}
